// OpenPanguMLAAttention_87711822119116
// MI455X (gfx1250) — matrix operand order repaired and checked by reading, not compiled, not run on hardware
//
#include <hip/hip_runtime.h>
#include <hip/hip_bf16.h>
#include <cstdint>

// ---------------------------------------------------------------------------
// Types for CDNA5 WMMA (wave32)
// ---------------------------------------------------------------------------
typedef __attribute__((ext_vector_type(16))) __bf16 bf16x16;
typedef __attribute__((ext_vector_type(8)))  __bf16 bf16x8;
typedef __attribute__((ext_vector_type(8)))  float  f32x8;
typedef int i32x4 __attribute__((vector_size(16)));   // matches async-LDS builtin param

#define T_TOK 2048
#define HID   2048
#define NH    16
#define DN    128
#define DR    64
#define DV    128
#define DQK   192      // DN + DR
#define QLR   1536
#define KVLR  512
#define QKV_N (QLR + KVLR + DR)   // 2112
#define ATT_SCALE 0.07216878364870323f   // (192)^-0.5
#define LN_THETA_OVER_HALF 0.28782313662425572f // ln(10000)/32

#if defined(__has_builtin)
#  if __has_builtin(__builtin_amdgcn_global_load_async_to_lds_b128)
#    define HAVE_ASYNC_LDS 1
#  endif
#endif
#ifndef HAVE_ASYNC_LDS
#  define HAVE_ASYNC_LDS 0
#endif

typedef __attribute__((address_space(1))) i32x4 gvec_t;   // global int4
typedef __attribute__((address_space(3))) i32x4 lvec_t;   // LDS int4

static __device__ __forceinline__ __bf16 f2bf(float f) { return (__bf16)f; }

static __device__ __forceinline__ void wait_lds() {
#if __has_builtin(__builtin_amdgcn_s_wait_dscnt)
    __builtin_amdgcn_s_wait_dscnt(0);
#else
    asm volatile("s_wait_dscnt 0" ::: "memory");
#endif
}

#if HAVE_ASYNC_LDS
static __device__ __forceinline__ void wait_async() {
#if __has_builtin(__builtin_amdgcn_s_wait_asynccnt)
    __builtin_amdgcn_s_wait_asynccnt(0);
#else
    asm volatile("s_wait_asynccnt 0" ::: "memory");
#endif
}
#endif

static __device__ __forceinline__ bf16x8 cvt8(float4 a, float4 b) {
    return (bf16x8){ (__bf16)a.x, (__bf16)a.y, (__bf16)a.z, (__bf16)a.w,
                     (__bf16)b.x, (__bf16)b.y, (__bf16)b.z, (__bf16)b.w };
}

// Load a 16x32 bf16 A-fragment from a row-major LDS tile (stride in elems).
// Layout: lane<16 -> row=lane, K 0..15; lane>=16 -> row=lane-16, K 16..31 (contiguous)
static __device__ __forceinline__ bf16x16 load_a_frag(const __bf16* base, int stride) {
    int lane  = threadIdx.x & 31;
    int m     = lane & 15;
    int khalf = lane >> 4;
    return *(const bf16x16*)(base + m * stride + khalf * 16);
}

// Load a 32x16 bf16 B-fragment from an N-major (transposed) LDS tile:
// lane<16 -> col=lane, K 0..15; lane>=16 -> col=lane-16, K 16..31 (contiguous)
static __device__ __forceinline__ bf16x16 load_b_frag(const __bf16* base, int stride) {
    int lane  = threadIdx.x & 31;
    int n     = lane & 15;
    int khalf = lane >> 4;
    return *(const bf16x16*)(base + n * stride + khalf * 16);
}

// ---------------------------------------------------------------------------
// GEMM: C[M,N] = A[M,K] * B[K,N], fp32 in/out, bf16 WMMA compute.
// Block = 256 threads (8 waves), tile 128(M) x 128(N), K-step 32.
// Wave (mw = wv&3, nw = wv>>2) computes rows [mw*32, +32) x cols [nw*64, +64)
// = 2x4 subtiles = 8 WMMAs per K-step.
// Requires M%128==0, K%32==0, N%4==0 (N edge tiles: loads clamped, stores guarded).
// ---------------------------------------------------------------------------
__global__ void __launch_bounds__(256)
gemm_bf16_wmma(const float* __restrict__ A, const float* __restrict__ B,
               float* __restrict__ C, int M, int N, int K) {
    __shared__ __attribute__((aligned(32))) __bf16 sA[128 * 32];   // row-major [m][k]
    __shared__ __attribute__((aligned(32))) __bf16 sB[128 * 32];   // N-major  [n][k]

    const int n0   = blockIdx.x * 128;
    const int m0   = blockIdx.y * 128;
    const int tid  = threadIdx.x;
    const int wv   = tid >> 5;
    const int lane = tid & 31;
    const int mw   = (wv & 3) * 32;
    const int nw   = (wv >> 2) * 64;

    f32x8 acc[2][4];
    #pragma unroll
    for (int i = 0; i < 2; ++i)
        #pragma unroll
        for (int j = 0; j < 4; ++j)
            acc[i][j] = (f32x8){0.f,0.f,0.f,0.f,0.f,0.f,0.f,0.f};

    // staging coordinates (fixed per thread)
    const int am = tid >> 1;          // A row 0..127
    const int ak = (tid & 1) * 16;    // A col 0 or 16

    for (int k0 = 0; k0 < K; k0 += 32) {
        __syncthreads();
        {   // Stage A tile 128x32: 16 floats per thread, vectorized
            const float4* src = (const float4*)(A + (size_t)(m0 + am) * K + k0 + ak);
            float4 f0 = src[0], f1 = src[1], f2 = src[2], f3 = src[3];
            *(bf16x8*)(sA + am * 32 + ak)     = cvt8(f0, f1);
            *(bf16x8*)(sA + am * 32 + ak + 8) = cvt8(f2, f3);
        }
        {   // Stage B tile 32x128 transposed -> sB[n][k]; branch-free clamped loads
            #pragma unroll
            for (int p = 0; p < 4; ++p) {
                int linear = tid + p * 256;          // 0..1023
                int k  = linear >> 5;                // 0..31
                int n4 = (linear & 31) * 4;          // 0..124
                int nc = (n0 + n4 < N) ? (n0 + n4) : (N - 4);  // clamp, never OOB
                float4 f = *(const float4*)(B + (size_t)(k0 + k) * N + nc);
                sB[(n4 + 0) * 32 + k] = f2bf(f.x);
                sB[(n4 + 1) * 32 + k] = f2bf(f.y);
                sB[(n4 + 2) * 32 + k] = f2bf(f.z);
                sB[(n4 + 3) * 32 + k] = f2bf(f.w);
            }
        }
        if (k0 + 32 < K)
            __builtin_prefetch(&A[(size_t)(m0 + am) * K + k0 + 32], 0, 1);
        __syncthreads();

        bf16x16 a0 = load_a_frag(sA + (mw +  0) * 32, 32);
        bf16x16 a1 = load_a_frag(sA + (mw + 16) * 32, 32);
        #pragma unroll
        for (int ns = 0; ns < 4; ++ns) {
            bf16x16 b = load_b_frag(sB + (nw + ns * 16) * 32, 32);
            acc[0][ns] = __builtin_amdgcn_wmma_f32_16x16x32_bf16(
                false, a0, false, b, (short)0, acc[0][ns], false, false);
            acc[1][ns] = __builtin_amdgcn_wmma_f32_16x16x32_bf16(
                false, a1, false, b, (short)0, acc[1][ns], false, false);
        }
    }

    // C layout: VGPR r -> row r (lanes 0-15) / row r+8 (lanes 16-31); col = lane&15
    const int rowoff = (lane & 16) ? 8 : 0;
    const int col    = lane & 15;
    #pragma unroll
    for (int sub = 0; sub < 2; ++sub) {
        #pragma unroll
        for (int ns = 0; ns < 4; ++ns) {
            #pragma unroll
            for (int r = 0; r < 8; ++r) {
                int row = m0 + mw + sub * 16 + r + rowoff;
                int c   = n0 + nw + ns * 16 + col;
                if (c < N) C[(size_t)row * N + c] = acc[sub][ns][r];
            }
        }
    }
}

// ---------------------------------------------------------------------------
// RMSNorm over rows: y[t, 0..C) = x[t*ld .. +C) * rsqrt(mean(x^2)+eps) * w
// ---------------------------------------------------------------------------
__global__ void __launch_bounds__(256)
rmsnorm_kernel(const float* __restrict__ x, int ld, int C,
               const float* __restrict__ w, float* __restrict__ y) {
    __shared__ float sbuf[256];
    const int t = blockIdx.x;
    const float* xr = x + (size_t)t * ld;
    float ss = 0.f;
    for (int i = threadIdx.x; i < C; i += 256) { float v = xr[i]; ss += v * v; }
    sbuf[threadIdx.x] = ss;
    __syncthreads();
    for (int s = 128; s > 0; s >>= 1) {
        if (threadIdx.x < s) sbuf[threadIdx.x] += sbuf[threadIdx.x + s];
        __syncthreads();
    }
    float r = rsqrtf(sbuf[0] / (float)C + 1e-6f);
    for (int i = threadIdx.x; i < C; i += 256)
        y[(size_t)t * C + i] = xr[i] * r * w[i];
}

// ---------------------------------------------------------------------------
// RoPE + pack into bf16 attention operands:
//   qf[t,h,0:192]         : q_nope | rope(q_pe)
//   kf[t,h,0:192]         : k_nope | rope(k_pe)  (k_pe broadcast over h)
//   vfT[t/32,h,dv,key32]  : V pre-transposed per 32-key block (async-copy ready)
// ---------------------------------------------------------------------------
__global__ void __launch_bounds__(128)
rope_pack_kernel(const float* __restrict__ qmat,   // [T, H*192]
                 const float* __restrict__ kvmat,  // [T, H*256]
                 const float* __restrict__ qkv_a,  // [T, 2112] (k_pe at col 2048)
                 const int*   __restrict__ pos,
                 __bf16* __restrict__ qf, __bf16* __restrict__ kf,
                 __bf16* __restrict__ vfT) {
    const int t = blockIdx.x;
    const float p = (float)pos[t];
    const int blk = t >> 5, key = t & 31;

    for (int idx = threadIdx.x; idx < NH * DQK; idx += 128) {
        int h = idx / DQK, d = idx % DQK;
        float qv, kv;
        if (d < DN) {
            qv = qmat[(size_t)t * (NH * DQK) + h * DQK + d];
            kv = kvmat[(size_t)t * (NH * (DN + DV)) + h * (DN + DV) + d];
        } else {
            int jj = d - DN, i = jj >> 1;
            float f = p * __expf(-(float)i * LN_THETA_OVER_HALF);
            float c = __cosf(f), s = __sinf(f);
            const float* qb = qmat + (size_t)t * (NH * DQK) + h * DQK + DN;
            float qx1 = qb[2 * i], qx2 = qb[2 * i + 1];
            const float* kb = qkv_a + (size_t)t * QKV_N + (QLR + KVLR);
            float kx1 = kb[2 * i], kx2 = kb[2 * i + 1];
            qv = (jj & 1) ? (qx1 * s + qx2 * c) : (qx1 * c - qx2 * s);
            kv = (jj & 1) ? (kx1 * s + kx2 * c) : (kx1 * c - kx2 * s);
        }
        qf[((size_t)t * NH + h) * DQK + d] = f2bf(qv);
        kf[((size_t)t * NH + h) * DQK + d] = f2bf(kv);
    }
    for (int idx = threadIdx.x; idx < NH * DV; idx += 128) {
        int h = idx / DV, d = idx % DV;
        vfT[(((size_t)blk * NH + h) * DV + d) * 32 + key] =
            f2bf(kvmat[(size_t)t * (NH * (DN + DV)) + h * (DN + DV) + DN + d]);
    }
}

// ---------------------------------------------------------------------------
// Causal flash attention, one head per blockIdx.y, 64 queries per block
// (4 waves x 16 rows). 32 keys per iteration. All matmuls via bf16 WMMA.
// V tile arrives via async global->LDS copy (contiguous, pre-transposed).
// ---------------------------------------------------------------------------
__global__ void __launch_bounds__(128)
mla_attention_kernel(const __bf16* __restrict__ qf, const __bf16* __restrict__ kf,
                     const __bf16* __restrict__ vfT, float* __restrict__ out) {
    __shared__ __attribute__((aligned(32))) __bf16 sVt[DV * 32];     // [dv][key] 8KB
    __shared__ __attribute__((aligned(32))) __bf16 sP[4][16 * 32];   // per-wave P 16x32

    const int h     = blockIdx.y;
    const int qb    = blockIdx.x;
    const int wv    = threadIdx.x >> 5;
    const int lane  = threadIdx.x & 31;
    const int laneN = lane & 15;
    const int khalf = lane >> 4;
    const int rowoff = (lane & 16) ? 8 : 0;
    const int qbase = qb * 64 + wv * 16;

    // Q fragments: 6 chunks of 16x32 (K = 192)
    bf16x16 qfrag[6];
    #pragma unroll
    for (int c = 0; c < 6; ++c) {
        const __bf16* p = qf + ((size_t)(qbase + laneN) * NH + h) * DQK + c * 32 + khalf * 16;
        qfrag[c] = *(const bf16x16*)(p);
    }

    float m_run[8], l_run[8];
    f32x8 o[8];
    #pragma unroll
    for (int r = 0; r < 8; ++r) { m_run[r] = -3.0e38f; l_run[r] = 0.f; }
    #pragma unroll
    for (int nb = 0; nb < 8; ++nb) o[nb] = (f32x8){0.f,0.f,0.f,0.f,0.f,0.f,0.f,0.f};

    const int qmax_block = qb * 64 + 63;
    for (int j = 0; j <= qmax_block; j += 32) {
        __syncthreads();
        {   // stage pre-transposed V block [dv][key]: contiguous 8KB global -> LDS
            char* vsrc = (char*)(vfT + (((size_t)(j >> 5) * NH + h) * DV) * 32);
            char* dst  = (char*)sVt;
#if HAVE_ASYNC_LDS
            #pragma unroll
            for (int it = 0; it < 4; ++it) {
                int off = (int)threadIdx.x * 16 + it * 2048;
                __builtin_amdgcn_global_load_async_to_lds_b128(
                    (gvec_t*)(vsrc + off), (lvec_t*)(dst + off), 0, 0);
            }
            wait_async();
#else
            #pragma unroll
            for (int it = 0; it < 4; ++it) {
                int off = (int)threadIdx.x * 16 + it * 2048;
                *(bf16x8*)(dst + off) = *(const bf16x8*)(vsrc + off);
            }
#endif
        }
        __syncthreads();

        // scores for two 16-key subtiles
        f32x8 s[2];
        #pragma unroll
        for (int sub = 0; sub < 2; ++sub) {
            f32x8 a = (f32x8){0.f,0.f,0.f,0.f,0.f,0.f,0.f,0.f};
            int key = j + sub * 16 + laneN;
            #pragma unroll
            for (int c = 0; c < 6; ++c) {
                // k_full row is already the N-major B layout: contiguous K chunk
                bf16x16 b = *(const bf16x16*)(kf + ((size_t)key * NH + h) * DQK
                                              + c * 32 + khalf * 16);
                a = __builtin_amdgcn_wmma_f32_16x16x32_bf16(
                    false, qfrag[c], false, b, (short)0, a, false, false);
            }
            s[sub] = a;
        }

        // online softmax per row (row values live across 16 lanes)
        #pragma unroll
        for (int r = 0; r < 8; ++r) {
            int qrow = qbase + r + rowoff;
            float s0 = s[0][r] * ATT_SCALE;
            float s1 = s[1][r] * ATT_SCALE;
            if (j + laneN > qrow)      s0 = -3.0e38f;
            if (j + 16 + laneN > qrow) s1 = -3.0e38f;
            float mx = fmaxf(s0, s1);
            #pragma unroll
            for (int off = 1; off < 16; off <<= 1)
                mx = fmaxf(mx, __shfl_xor(mx, off, 32));
            float mnew = fmaxf(m_run[r], mx);
            float p0 = __expf(s0 - mnew);
            float p1 = __expf(s1 - mnew);
            float rs = p0 + p1;
            #pragma unroll
            for (int off = 1; off < 16; off <<= 1)
                rs += __shfl_xor(rs, off, 32);
            float alpha = __expf(m_run[r] - mnew);
            l_run[r] = l_run[r] * alpha + rs;
            m_run[r] = mnew;
            #pragma unroll
            for (int nb = 0; nb < 8; ++nb) o[nb][r] *= alpha;
            sP[wv][(r + rowoff) * 32 + laneN]      = f2bf(p0);
            sP[wv][(r + rowoff) * 32 + 16 + laneN] = f2bf(p1);
        }
        wait_lds();   // same-wave LDS write -> read turnaround

        // O += P (16x32) * V (32x128)
        bf16x16 pa = load_a_frag(sP[wv], 32);
        #pragma unroll
        for (int nb = 0; nb < 8; ++nb) {
            bf16x16 bv = load_b_frag(sVt + nb * 16 * 32, 32);
            o[nb] = __builtin_amdgcn_wmma_f32_16x16x32_bf16(
                false, pa, false, bv, (short)0, o[nb], false, false);
        }
    }

    #pragma unroll
    for (int nb = 0; nb < 8; ++nb) {
        #pragma unroll
        for (int r = 0; r < 8; ++r) {
            int qrow = qbase + r + rowoff;
            out[(size_t)qrow * (NH * DV) + h * DV + nb * 16 + laneN] =
                o[nb][r] / l_run[r];
        }
    }
}

// ---------------------------------------------------------------------------
// Host-side orchestration
// ---------------------------------------------------------------------------
extern "C" void kernel_launch(void* const* d_in, const int* in_sizes, int n_in,
                              void* d_out, int out_size, void* d_ws, size_t ws_size,
                              hipStream_t stream) {
    (void)in_sizes; (void)n_in; (void)out_size; (void)ws_size;

    const int*   positions = (const int*)  d_in[0];
    const float* hidden    = (const float*)d_in[1];
    const float* Wfused    = (const float*)d_in[2];   // [2048, 2112]
    const float* q_norm_w  = (const float*)d_in[3];   // [1536]
    const float* W_q_b     = (const float*)d_in[4];   // [1536, 3072]
    const float* kv_norm_w = (const float*)d_in[5];   // [512]
    const float* W_kv_b    = (const float*)d_in[6];   // [512, 4096]
    const float* W_o       = (const float*)d_in[7];   // [2048, 2048]

    char* w = (char*)d_ws;
    float* qkv_a    = (float*)w;  w += (size_t)T_TOK * QKV_N * 4;            // 17.3 MB
    float* qn       = (float*)w;  w += (size_t)T_TOK * QLR * 4;              // 12.6 MB
    float* kvn      = (float*)w;  w += (size_t)T_TOK * KVLR * 4;             //  4.2 MB
    float* qmat     = (float*)w;  w += (size_t)T_TOK * NH * DQK * 4;         // 25.2 MB
    float* kvmat    = (float*)w;  w += (size_t)T_TOK * NH * (DN + DV) * 4;   // 33.6 MB
    __bf16* qfb     = (__bf16*)w; w += (size_t)T_TOK * NH * DQK * 2;         // 12.6 MB
    __bf16* kfb     = (__bf16*)w; w += (size_t)T_TOK * NH * DQK * 2;         // 12.6 MB
    __bf16* vfT     = (__bf16*)w; w += (size_t)T_TOK * NH * DV * 2;          //  8.4 MB
    float* attn_out = (float*)w;  w += (size_t)T_TOK * HID * 4;              // 16.8 MB

    // 1) qkv_a = hidden @ W_fused_qkv_a   [2048x2048]x[2048x2112]
    gemm_bf16_wmma<<<dim3((QKV_N + 127) / 128, T_TOK / 128), 256, 0, stream>>>(
        hidden, Wfused, qkv_a, T_TOK, QKV_N, HID);

    // 2) RMSNorm slices of qkv_a
    rmsnorm_kernel<<<T_TOK, 256, 0, stream>>>(qkv_a, QKV_N, QLR, q_norm_w, qn);
    rmsnorm_kernel<<<T_TOK, 256, 0, stream>>>(qkv_a + QLR, QKV_N, KVLR, kv_norm_w, kvn);

    // 3) q = qn @ W_q_b    [2048x1536]x[1536x3072]
    gemm_bf16_wmma<<<dim3((NH * DQK) / 128, T_TOK / 128), 256, 0, stream>>>(
        qn, W_q_b, qmat, T_TOK, NH * DQK, QLR);

    // 4) kv = kvn @ W_kv_b [2048x512]x[512x4096]
    gemm_bf16_wmma<<<dim3((NH * (DN + DV)) / 128, T_TOK / 128), 256, 0, stream>>>(
        kvn, W_kv_b, kvmat, T_TOK, NH * (DN + DV), KVLR);

    // 5) RoPE + pack to bf16 attention operands (V pre-transposed per 32-key block)
    rope_pack_kernel<<<T_TOK, 128, 0, stream>>>(
        qmat, kvmat, qkv_a, positions, qfb, kfb, vfT);

    // 6) causal flash attention
    mla_attention_kernel<<<dim3(T_TOK / 64, NH), 128, 0, stream>>>(
        qfb, kfb, vfT, attn_out);

    // 7) out = attn_out @ W_o  [2048x2048]x[2048x2048]
    gemm_bf16_wmma<<<dim3(HID / 128, T_TOK / 128), 256, 0, stream>>>(
        attn_out, W_o, (float*)d_out, T_TOK, HID, HID);
}
